// CausalMHA_71270687310219
// MI455X (gfx1250) — matrix operand order repaired and checked by reading, not compiled, not run on hardware
//
#include <hip/hip_runtime.h>

typedef __attribute__((ext_vector_type(16))) __bf16 v16bf;
typedef __attribute__((ext_vector_type(8)))  float  v8f;

constexpr int Bsz = 4, Seq = 2048, Dm = 1024, Hh = 16, Eh = 64;
constexpr int Mtot = Bsz * Seq; // 8192

#if __has_builtin(__builtin_amdgcn_global_load_async_to_lds_b128)
#define HAVE_ASYNC_LDS 1
#else
#define HAVE_ASYNC_LDS 0
#endif

#if __has_builtin(__builtin_amdgcn_permlane16)
#define HAVE_PERMLANE16 1
#else
#define HAVE_PERMLANE16 0
#endif

// vector-of-4-int pointer types matching the async builtin's signature
typedef int v4i_vs __attribute__((vector_size(16)));
typedef __attribute__((address_space(1))) v4i_vs* gvp4;
typedef __attribute__((address_space(3))) v4i_vs* lvp4;

// 16-byte global -> LDS copy. CDNA5 async-DMA path (ASYNCcnt, bypasses VGPRs).
__device__ __forceinline__ void async_cp16(void* lds, const void* g) {
#if HAVE_ASYNC_LDS
    __builtin_amdgcn_global_load_async_to_lds_b128((gvp4)(void*)g, (lvp4)lds, 0, 0);
#else
    *(uint4*)lds = *(const uint4*)g;
#endif
}

__device__ __forceinline__ void wait_async0() {
#if HAVE_ASYNC_LDS
#if __has_builtin(__builtin_amdgcn_s_wait_asynccnt)
    __builtin_amdgcn_s_wait_asynccnt(0);
#else
    asm volatile("s_wait_asynccnt 0x0" ::: "memory");
#endif
#endif
}

// max over each 16-lane half (rows of a WMMA C tile) via VALU permlane16
// xor-butterfly; LDS-permute fallback.
__device__ __forceinline__ float rowmax16(float x) {
#if HAVE_PERMLANE16
    const unsigned lo[4] = {0x67452301u, 0x45670123u, 0x01234567u, 0xFEDCBA98u};
    const unsigned hi[4] = {0xEFCDAB89u, 0xCDEF89ABu, 0x89ABCDEFu, 0x76543210u};
#pragma unroll
    for (int s = 0; s < 4; s++) {
        union { float f; int i; } v; v.f = x;
        union { int i; float f; } w;
        w.i = __builtin_amdgcn_permlane16(v.i, v.i, lo[s], hi[s], false, false);
        x = fmaxf(x, w.f);
    }
#else
#pragma unroll
    for (int off = 1; off < 16; off <<= 1)
        x = fmaxf(x, __shfl_xor(x, off, 32));
#endif
    return x;
}

__device__ __forceinline__ unsigned short f2bf(float x) {
    union { float f; unsigned u; } v; v.f = x;
    unsigned r = v.u + 0x7FFFu + ((v.u >> 16) & 1u);
    return (unsigned short)(r >> 16);
}

__device__ __forceinline__ v8f wmma_bf16(v16bf a, v16bf b, v8f c) {
    return __builtin_amdgcn_wmma_f32_16x16x32_bf16(
        false, a, false, b, (short)0, c, false, false);
}

union FragU { uint4 u[2]; v16bf v; };

// pack 16 fp32 -> 16 bf16 (rne) into LDS
__device__ __forceinline__ void pack16(unsigned short* dst, float4 a, float4 b,
                                       float4 c, float4 d) {
    float fv[16] = {a.x, a.y, a.z, a.w, b.x, b.y, b.z, b.w,
                    c.x, c.y, c.z, c.w, d.x, d.y, d.z, d.w};
    unsigned* u = (unsigned*)dst;
#pragma unroll
    for (int i = 0; i < 8; i++)
        u[i] = (unsigned)f2bf(fv[2 * i]) | ((unsigned)f2bf(fv[2 * i + 1]) << 16);
}

// ---------------------------------------------------------------------------
// GEMM: C[M=8192, N=1024] = outScale * (A[M, K=1024] * W^T)
// Double-buffered LDS; bf16 A staged via async-to-LDS, fp32 via reg prefetch.
// ---------------------------------------------------------------------------
template <bool A_BF16, bool OUT_BF16, bool TRANS_V>
__launch_bounds__(256)
__global__ void gemm_xwt_kernel(const void* __restrict__ Ap,
                                const float* __restrict__ W,
                                void* __restrict__ Cp, float outScale) {
    constexpr int Kd = Dm, Nd = Dm;
    __shared__ unsigned short As[2][128][40];
    __shared__ unsigned short Ws[2][128][40];

    const int t = threadIdx.x, lane = t & 31, wave = t >> 5;
    const int wm = wave >> 2, wn = wave & 3; // 2 x 4 wave grid
    const int mblk = blockIdx.y * 128, nblk = blockIdx.x * 128;
    const int hi = lane >> 4, r16 = lane & 15;

    v8f acc[4][2];
#pragma unroll
    for (int i = 0; i < 4; i++)
#pragma unroll
        for (int j = 0; j < 2; j++)
#pragma unroll
            for (int r = 0; r < 8; r++) acc[i][j][r] = 0.0f;

    const int lrow = t >> 1, lch = (t & 1) * 16;
    const float* Af = (const float*)Ap;
    const unsigned short* Ab = (const unsigned short*)Ap;

    float4 ra0, ra1, ra2, ra3, rw0, rw1, rw2, rw3;

    auto loadW = [&](int k0) {
        const float4* p = (const float4*)(W + (size_t)(nblk + lrow) * Kd + k0 + lch);
        rw0 = p[0]; rw1 = p[1]; rw2 = p[2]; rw3 = p[3];
    };
    auto storeW = [&](int bu) { pack16(&Ws[bu][lrow][lch], rw0, rw1, rw2, rw3); };
    auto loadA = [&](int k0) {
        const float4* p = (const float4*)(Af + (size_t)(mblk + lrow) * Kd + k0 + lch);
        ra0 = p[0]; ra1 = p[1]; ra2 = p[2]; ra3 = p[3];
    };
    auto storeA = [&](int bu) { pack16(&As[bu][lrow][lch], ra0, ra1, ra2, ra3); };
    auto asyncA = [&](int k0, int bu) {
        const unsigned short* g = Ab + (size_t)(mblk + lrow) * Kd + k0 + lch;
        async_cp16(&As[bu][lrow][lch], g);
        async_cp16(&As[bu][lrow][lch + 8], g + 8);
    };

    // ---- prologue: stage tile 0 into buffer 0 ----
    if constexpr (A_BF16) asyncA(0, 0); else loadA(0);
    loadW(0);
    if constexpr (!A_BF16) storeA(0);
    storeW(0);
    wait_async0();
    __syncthreads();

    int buf = 0;
    for (int k0 = 0; k0 < Kd; k0 += 32) {
        const bool more = (k0 + 32) < Kd;
        if (more) { // issue next-tile traffic before the math
            if constexpr (A_BF16) asyncA(k0 + 32, buf ^ 1); else loadA(k0 + 32);
            loadW(k0 + 32);
            if (k0 + 64 < Kd)
                __builtin_prefetch(W + (size_t)(nblk + lrow) * Kd + k0 + 64 + lch, 0, 1);
        }

        // ---- fragments + WMMA on current buffer ----
        FragU af[4], bfr[2];
        const int ak = hi ? 8 : 0, bk = hi ? 8 : 0;
#pragma unroll
        for (int i = 0; i < 4; i++) {
            const unsigned short* pA = &As[buf][wm * 64 + i * 16 + r16][ak];
            af[i].u[0] = *(const uint4*)pA;
            af[i].u[1] = *(const uint4*)(pA + 16);
        }
#pragma unroll
        for (int j = 0; j < 2; j++) {
            const unsigned short* pB = &Ws[buf][wn * 32 + j * 16 + r16][bk];
            bfr[j].u[0] = *(const uint4*)pB;
            bfr[j].u[1] = *(const uint4*)(pB + 16);
        }
#pragma unroll
        for (int i = 0; i < 4; i++)
#pragma unroll
            for (int j = 0; j < 2; j++)
                acc[i][j] = wmma_bf16(af[i].v, bfr[j].v, acc[i][j]);

        if (more) { // commit next tile into back buffer, single barrier
            if constexpr (!A_BF16) storeA(buf ^ 1);
            storeW(buf ^ 1);
            wait_async0();
            __syncthreads();
            buf ^= 1;
        }
    }

    // ---- epilogue ----
#pragma unroll
    for (int i = 0; i < 4; i++)
#pragma unroll
        for (int j = 0; j < 2; j++)
#pragma unroll
            for (int r = 0; r < 8; r++) {
                int grow = mblk + wm * 64 + i * 16 + r + hi * 8;
                int gcol = nblk + wn * 32 + j * 16 + r16;
                float val = acc[i][j][r] * outScale;
                if constexpr (OUT_BF16) {
                    unsigned short* C = (unsigned short*)Cp;
                    size_t off;
                    if constexpr (TRANS_V) {
                        int bb = grow >> 11, s = grow & (Seq - 1);
                        off = (size_t)(bb * Dm + gcol) * Seq + s;
                    } else {
                        off = (size_t)grow * Nd + gcol;
                    }
                    C[off] = f2bf(val);
                } else {
                    ((float*)Cp)[(size_t)grow * Nd + gcol] = val;
                }
            }
}

// ---------------------------------------------------------------------------
// Flash attention: per (b, h, 128-row q-tile). 8 waves x 16 q-rows.
// Q is pre-scaled by 1/sqrt(E) at projection time. Double-buffered async
// K/V staging; row-sum via WMMA against all-ones B; masking only on
// diagonal-straddling tiles.
// ---------------------------------------------------------------------------
__launch_bounds__(256)
__global__ void flash_attn_kernel(const unsigned short* __restrict__ Q,
                                  const unsigned short* __restrict__ Kw,
                                  const unsigned short* __restrict__ Vt,
                                  unsigned short* __restrict__ Aout) {
    __shared__ unsigned short Qs[128][72];
    __shared__ unsigned short Ks[2][32][72];
    __shared__ unsigned short Vs[2][64][40];
    __shared__ unsigned short Ps[8][16][40];

    const int t = threadIdx.x, lane = t & 31, wave = t >> 5;
    const int qt = blockIdx.x, h = blockIdx.y, b = blockIdx.z;
    const int hi = lane >> 4, r16 = lane & 15;
    const int q0 = qt * 128;

    auto stageK = [&](int kc, int bu) {
        int row = t >> 3, ch = (t & 7) * 8;
        async_cp16(&Ks[bu][row][ch],
                   Kw + ((size_t)(b * Seq + kc + row) * Dm + h * Eh + ch));
    };
    auto stageV = [&](int kc, int bu) {
        int erow = t >> 2, ch = (t & 3) * 8;
        async_cp16(&Vs[bu][erow][ch],
                   Vt + ((size_t)(b * Dm + h * Eh + erow) * Seq + kc + ch));
    };

    // ---- prologue: Q tile + first K/V tiles ----
    {
        int row = t >> 1, ch = (t & 1) * 32;
        const unsigned short* g = Q + ((size_t)(b * Seq + q0 + row) * Dm + h * Eh + ch);
#pragma unroll
        for (int i = 0; i < 4; i++)
            async_cp16(&Qs[row][ch + i * 8], g + i * 8);
    }
    stageK(0, 0);
    stageV(0, 0);
    wait_async0();
    __syncthreads();

    FragU qf[2];
#pragma unroll
    for (int ec = 0; ec < 2; ec++) {
        const unsigned short* p = &Qs[wave * 16 + r16][ec * 32 + (hi ? 8 : 0)];
        qf[ec].u[0] = *(const uint4*)p;
        qf[ec].u[1] = *(const uint4*)(p + 16);
    }

    FragU ones;
    ones.u[0] = ones.u[1] =
        make_uint4(0x3F803F80u, 0x3F803F80u, 0x3F803F80u, 0x3F803F80u);

    float mstate[8];
    v8f oacc[4], lacc;
#pragma unroll
    for (int r = 0; r < 8; r++) { mstate[r] = -1e30f; lacc[r] = 0.0f; }
#pragma unroll
    for (int jn = 0; jn < 4; jn++)
#pragma unroll
        for (int r = 0; r < 8; r++) oacc[jn][r] = 0.0f;

    const int qlo  = q0 + wave * 16;      // smallest q-row this wave owns
    const int qmax = qlo + 15;            // largest
    const int kcend = q0 + 128;

    int buf = 0;
    for (int kc = 0; kc < kcend; kc += 32) {
        const int kn = kc + 32;
        if (kn < kcend) { // prefetch next K/V tile into back buffer
            stageK(kn, buf ^ 1);
            stageV(kn, buf ^ 1);
            if (kn + 32 < kcend)
                __builtin_prefetch(
                    Kw + ((size_t)(b * Seq + kn + 32 + (t >> 3)) * Dm + h * Eh), 0, 1);
        }

        if (kc <= qmax) { // wave-uniform causal skip
            // ---- S = Q K^T (Q pre-scaled) ----
            v8f sc[2];
#pragma unroll
            for (int j = 0; j < 2; j++)
#pragma unroll
                for (int r = 0; r < 8; r++) sc[j][r] = 0.0f;
#pragma unroll
            for (int j = 0; j < 2; j++)
#pragma unroll
                for (int ec = 0; ec < 2; ec++) {
                    FragU kf;
                    const unsigned short* p =
                        &Ks[buf][j * 16 + r16][ec * 32 + (hi ? 8 : 0)];
                    kf.u[0] = *(const uint4*)p;
                    kf.u[1] = *(const uint4*)(p + 16);
                    sc[j] = wmma_bf16(qf[ec].v, kf.v, sc[j]);
                }
            // ---- causal mask only on diagonal-straddling tiles ----
            if (kc + 31 > qlo) { // wave-uniform
#pragma unroll
                for (int j = 0; j < 2; j++)
#pragma unroll
                    for (int r = 0; r < 8; r++) {
                        int kidx = kc + j * 16 + r16;
                        int qidx = qlo + r + hi * 8;
                        if (kidx > qidx) sc[j][r] = -1e30f;
                    }
            }
            // ---- online softmax: permlane butterfly for row-max only ----
            float pnew[2][8];
#pragma unroll
            for (int r = 0; r < 8; r++) {
                float tmax = rowmax16(fmaxf(sc[0][r], sc[1][r]));
                float mnew  = fmaxf(mstate[r], tmax);
                float alpha = __expf(mstate[r] - mnew);
                pnew[0][r] = __expf(sc[0][r] - mnew);
                pnew[1][r] = __expf(sc[1][r] - mnew);
                mstate[r] = mnew;
                lacc[r] *= alpha;
#pragma unroll
                for (int jn = 0; jn < 4; jn++) oacc[jn][r] *= alpha;
            }
            // ---- P (C-layout) -> LDS -> A-layout fragment ----
#pragma unroll
            for (int j = 0; j < 2; j++)
#pragma unroll
                for (int r = 0; r < 8; r++)
                    Ps[wave][r + hi * 8][j * 16 + r16] = f2bf(pnew[j][r]);
            asm volatile("s_wait_dscnt 0x0" ::: "memory");
            FragU pf;
            {
                const unsigned short* pp = &Ps[wave][r16][hi ? 8 : 0];
                pf.u[0] = *(const uint4*)pp;
                pf.u[1] = *(const uint4*)(pp + 16);
            }
            // ---- l += rowsum(P) via WMMA against ones ----
            lacc = wmma_bf16(pf.v, ones.v, lacc);
            // ---- O += P V ----
#pragma unroll
            for (int jn = 0; jn < 4; jn++) {
                FragU vf;
                const unsigned short* pv = &Vs[buf][jn * 16 + r16][hi ? 8 : 0];
                vf.u[0] = *(const uint4*)pv;
                vf.u[1] = *(const uint4*)(pv + 16);
                oacc[jn] = wmma_bf16(pf.v, vf.v, oacc[jn]);
            }
        }
        wait_async0();
        __syncthreads();
        buf ^= 1;
    }

    // ---- finalize: O / l, store bf16 as [B,S,D] ----
#pragma unroll
    for (int r = 0; r < 8; r++) {
        float inv = 1.0f / lacc[r];
        int srow = qlo + r + hi * 8;
#pragma unroll
        for (int jn = 0; jn < 4; jn++) {
            int col = h * Eh + jn * 16 + r16;
            Aout[(size_t)(b * Seq + srow) * Dm + col] = f2bf(oacc[jn][r] * inv);
        }
    }
}

// ---------------------------------------------------------------------------
extern "C" void kernel_launch(void* const* d_in, const int* in_sizes, int n_in,
                              void* d_out, int out_size, void* d_ws, size_t ws_size,
                              hipStream_t stream) {
    const float* X  = (const float*)d_in[0];
    const float* Wq = (const float*)d_in[1];
    const float* Wk = (const float*)d_in[2];
    const float* Wv = (const float*)d_in[3];
    const float* Wo = (const float*)d_in[4];
    float* out = (float*)d_out;

    size_t elems = (size_t)Mtot * Dm; // 8,388,608 bf16 per tensor
    unsigned short* qws = (unsigned short*)d_ws;
    unsigned short* kws = qws + elems;
    unsigned short* vws = kws + elems;
    unsigned short* aws = vws + elems;

    dim3 gg(Dm / 128, Mtot / 128); // (8, 64)
    // Q is pre-scaled by 1/sqrt(E) so attention applies no per-score scale.
    gemm_xwt_kernel<false, true, false><<<gg, 256, 0, stream>>>(X, Wq, qws, 0.125f);
    gemm_xwt_kernel<false, true, false><<<gg, 256, 0, stream>>>(X, Wk, kws, 1.0f);
    gemm_xwt_kernel<false, true, true ><<<gg, 256, 0, stream>>>(X, Wv, vws, 1.0f);

    flash_attn_kernel<<<dim3(Seq / 128, Hh, Bsz), 256, 0, stream>>>(qws, kws, vws, aws);

    gemm_xwt_kernel<true, false, false><<<gg, 256, 0, stream>>>(aws, Wo, out, 1.0f);
}
